// SelfAttention_18777597018329
// MI455X (gfx1250) — matrix operand order repaired and checked by reading, not compiled, not run on hardware
//
#include <hip/hip_runtime.h>
#include <hip/hip_bf16.h>

typedef __bf16 bf16;
typedef __attribute__((ext_vector_type(16))) __bf16 v16bf;
typedef __attribute__((ext_vector_type(8)))  __bf16 v8bf;
typedef __attribute__((ext_vector_type(8)))  float  v8f;
typedef __attribute__((address_space(3))) bf16 lds_bf16;

__device__ __forceinline__ v8f zero8() {
    v8f z;
#pragma unroll
    for (int i = 0; i < 8; ++i) z[i] = 0.0f;
    return z;
}

__device__ __forceinline__ v8f wmma_bf16(v16bf a, v16bf b, v8f c) {
    // D = A(16x32 bf16) * B(32x16 bf16) + C(16x16 f32)
    return __builtin_amdgcn_wmma_f32_16x16x32_bf16(
        /*neg_a=*/false, a, /*neg_b=*/false, b,
        /*c_mod=*/(short)0, c, /*reuse_a=*/false, /*reuse_b=*/false);
}

// Async global -> LDS copy: 16 bytes per lane, tracked by ASYNCcnt.
__device__ __forceinline__ void async_b128(void* lds_dst, const void* gsrc) {
    lds_bf16* lp = (lds_bf16*)lds_dst;
    asm volatile("global_load_async_to_lds_b128 %0, %1, off"
                 :: "v"(lp), "v"(gsrc) : "memory");
}
__device__ __forceinline__ void wait_async0() {
    asm volatile("s_wait_asynccnt 0x0" ::: "memory");
}

// ---------------------------------------------------------------------------
// Kernel 0: transpose + fp32->bf16 convert of a weight matrix.
// src: [rows, cols] f32 ; dst: [cols, rows] bf16
// ---------------------------------------------------------------------------
__global__ __launch_bounds__(256) void k_transpose_bf16(
    const float* __restrict__ src, bf16* __restrict__ dst, int rows, int cols)
{
    int idx = blockIdx.x * 256 + threadIdx.x;
    if (idx >= rows * cols) return;
    int r = idx / cols, c = idx % cols;
    dst[(size_t)c * rows + r] = (bf16)src[idx];
}

// ---------------------------------------------------------------------------
// Kernel 1: QKV projection.  C[8192,3072] = X[8192,1024] @ Wqkv + bias
// Block = 8 waves -> 128x64 tile; each wave a 16x64 strip.
// Weight tile (shared by all 8 waves) is double-buffered in LDS via async DMA.
// Writes Q (scaled by 1/8) / K as [BH,S,64] bf16, V transposed as [BH,64,S].
// ---------------------------------------------------------------------------
__global__ __launch_bounds__(256) void k_qkv_gemm(
    const float* __restrict__ X, const bf16* __restrict__ WT,
    const float* __restrict__ bias,
    bf16* __restrict__ Qb, bf16* __restrict__ Kb, bf16* __restrict__ VTb)
{
    __shared__ alignas(64) bf16 lds_b[2][64][32];   // [buf][n][k] 4KB per buf

    const int tid = threadIdx.x;
    const int lane = tid & 31;
    const int wave = tid >> 5;
    const int g = lane >> 4, r = lane & 15;
    const int row0 = blockIdx.y * 128 + wave * 16;
    const int col0 = blockIdx.x * 64;

    // staging pattern: thread -> (n = tid/4, k-offset = (tid%4)*8), 16B each
    const int bn = tid >> 2;
    const int bk = (tid & 3) * 8;
    const bf16* wsrc = WT + (size_t)(col0 + bn) * 1024 + bk;

    v8f acc[4];
#pragma unroll
    for (int t = 0; t < 4; ++t) acc[t] = zero8();

    const float* xrow = X + (size_t)(row0 + r) * 1024;

    async_b128(&lds_b[0][bn][bk], wsrc);
    for (int k0 = 0; k0 < 1024; k0 += 32) {
        const int buf = (k0 >> 5) & 1;
        wait_async0();
        __syncthreads();
        if (k0 + 32 < 1024) async_b128(&lds_b[buf ^ 1][bn][bk], wsrc + k0 + 32);

        __builtin_prefetch(xrow + k0 + 64, 0, 1);
        // A tile, 16x32 bf16, contiguous layout (lanes 0-15: K 0-15; 16-31: K 16-31)
        v8f lo = *(const v8f*)(xrow + k0 + 16 * g);
        v8f hi = *(const v8f*)(xrow + k0 + 16 * g + 8);
        v16bf a;
#pragma unroll
        for (int i = 0; i < 8; ++i) { a[i] = (bf16)lo[i]; a[8 + i] = (bf16)hi[i]; }
#pragma unroll
        for (int t = 0; t < 4; ++t) {
            v16bf b = *(const v16bf*)&lds_b[buf][16 * t + r][16 * g];
            acc[t] = wmma_bf16(a, b, acc[t]);
        }
    }

    // Epilogue: bias + scatter into attention layouts
#pragma unroll
    for (int t = 0; t < 4; ++t) {
        int col = col0 + 16 * t + r;
        float bv = bias[col];
        int which = col >> 10;
        int h = (col >> 6) & 15;
        int dh = col & 63;
#pragma unroll
        for (int rr = 0; rr < 8; ++rr) {
            int row = row0 + rr + 8 * g;      // C layout: M = rr + 8*g, N = r
            int b_ = row >> 11, s = row & 2047;
            float v = acc[t][rr] + bv;
            size_t bh = (size_t)(b_ * 16 + h);
            if (which == 0)      Qb[(bh * 2048 + s) * 64 + dh] = (bf16)(v * 0.125f);
            else if (which == 1) Kb[(bh * 2048 + s) * 64 + dh] = (bf16)v;
            else                 VTb[(bh * 64 + dh) * 2048 + s] = (bf16)v;
        }
    }
}

// ---------------------------------------------------------------------------
// Kernel 2: flash attention. One wave = 16 query rows; 8 waves share each
// 32-key K/V chunk via async-DMA double-buffered LDS tiles.
// grid = (S/128, B*H), block = 256.
// ---------------------------------------------------------------------------
__global__ __launch_bounds__(256) void k_attention(
    const bf16* __restrict__ Qb, const bf16* __restrict__ Kb,
    const bf16* __restrict__ VTb, bf16* __restrict__ AO)
{
    __shared__ alignas(64) bf16 lds_k[2][32][64];   // [buf][key][hd]
    __shared__ alignas(64) bf16 lds_v[2][64][32];   // [buf][hd][key]
    __shared__ alignas(64) bf16 lds_p[8][16][32];   // per-wave P C->A transpose

    const int tid = threadIdx.x;
    const int lane = tid & 31;
    const int wave = tid >> 5;
    const int g = lane >> 4, r = lane & 15;
    const int bh = blockIdx.y;
    const int q0 = blockIdx.x * 128 + wave * 16;

    const bf16* Kbase = Kb + (size_t)bh * 2048 * 64;
    const bf16* Vbase = VTb + (size_t)bh * 64 * 2048;

    // staging: K tile 32x64 (thread -> key=tid/8, hd=(tid%8)*8)
    //          V tile 64x32 (thread -> row=tid/4, kk=(tid%4)*8)
    const int kkey = tid >> 3, khd = (tid & 7) * 8;
    const int vrow = tid >> 2, vkk = (tid & 3) * 8;

    // Q tile: A-layout for head-dim chunks [0,32) and [32,64)
    const bf16* qrow = Qb + ((size_t)bh * 2048 + q0 + r) * 64;
    v16bf aq0, aq1;
    {
        v8bf l0 = *(const v8bf*)(qrow + 16 * g);
        v8bf h0 = *(const v8bf*)(qrow + 8 + 16 * g);
        v8bf l1 = *(const v8bf*)(qrow + 32 + 16 * g);
        v8bf h1 = *(const v8bf*)(qrow + 40 + 16 * g);
#pragma unroll
        for (int i = 0; i < 8; ++i) {
            aq0[i] = l0[i]; aq0[8 + i] = h0[i];
            aq1[i] = l1[i]; aq1[8 + i] = h1[i];
        }
    }

    v8f o[4];
#pragma unroll
    for (int t = 0; t < 4; ++t) o[t] = zero8();
    float m[8], l[8];
#pragma unroll
    for (int i = 0; i < 8; ++i) { m[i] = -1e30f; l[i] = 0.0f; }
    const float LOG2E = 1.4426950408889634f;

    // prime buffer 0
    async_b128(&lds_k[0][kkey][khd], Kbase + (size_t)kkey * 64 + khd);
    async_b128(&lds_v[0][vrow][vkk], Vbase + (size_t)vrow * 2048 + vkk);

    for (int key0 = 0; key0 < 2048; key0 += 32) {
        const int buf = (key0 >> 5) & 1;
        wait_async0();
        __syncthreads();
        if (key0 + 32 < 2048) {
            int kn = key0 + 32;
            async_b128(&lds_k[buf ^ 1][kkey][khd], Kbase + (size_t)(kn + kkey) * 64 + khd);
            async_b128(&lds_v[buf ^ 1][vrow][vkk], Vbase + (size_t)vrow * 2048 + kn + vkk);
        }

        // ---- scores: two 16x16 tiles over the 32-key chunk (head dim 64 = 2x32)
        v8f s0 = zero8(), s1 = zero8();
        {
            const bf16* kr = &lds_k[buf][r][0];
            v16bf b0 = *(const v16bf*)(kr + 16 * g);
            v16bf b1 = *(const v16bf*)(kr + 32 + 16 * g);
            s0 = wmma_bf16(aq0, b0, s0);
            s0 = wmma_bf16(aq1, b1, s0);
        }
        {
            const bf16* kr = &lds_k[buf][16 + r][0];
            v16bf b0 = *(const v16bf*)(kr + 16 * g);
            v16bf b1 = *(const v16bf*)(kr + 32 + 16 * g);
            s1 = wmma_bf16(aq0, b0, s1);
            s1 = wmma_bf16(aq1, b1, s1);
        }

        // ---- online softmax (row = rr + 8g, cols striped across 16-lane group)
        float scale[8];
#pragma unroll
        for (int rr = 0; rr < 8; ++rr) {
            float vmax = fmaxf(s0[rr], s1[rr]);
#pragma unroll
            for (int msk = 1; msk < 16; msk <<= 1)
                vmax = fmaxf(vmax, __shfl_xor(vmax, msk, 32));
            float mn = fmaxf(m[rr], vmax);
            float sc = __builtin_amdgcn_exp2f((m[rr] - mn) * LOG2E);
            float p0 = __builtin_amdgcn_exp2f((s0[rr] - mn) * LOG2E);
            float p1 = __builtin_amdgcn_exp2f((s1[rr] - mn) * LOG2E);
            float rs = p0 + p1;
#pragma unroll
            for (int msk = 1; msk < 16; msk <<= 1)
                rs += __shfl_xor(rs, msk, 32);
            l[rr] = l[rr] * sc + rs;
            m[rr] = mn;
            scale[rr] = sc;
            s0[rr] = p0; s1[rr] = p1;
        }
#pragma unroll
        for (int t = 0; t < 4; ++t)
#pragma unroll
            for (int rr = 0; rr < 8; ++rr) o[t][rr] *= scale[rr];

        // ---- P: C-layout -> A-layout via per-wave LDS tile
#pragma unroll
        for (int rr = 0; rr < 8; ++rr) {
            lds_p[wave][rr + 8 * g][r]      = (bf16)s0[rr];
            lds_p[wave][rr + 8 * g][r + 16] = (bf16)s1[rr];
        }
        asm volatile("s_wait_dscnt 0" ::: "memory");
        v16bf ap;
        {
            const bf16* prow = &lds_p[wave][r][0];
            v8bf pl = *(const v8bf*)(prow + 16 * g);
            v8bf ph = *(const v8bf*)(prow + 8 + 16 * g);
#pragma unroll
            for (int i = 0; i < 8; ++i) { ap[i] = pl[i]; ap[8 + i] = ph[i]; }
        }
        asm volatile("" ::: "memory");

        // ---- PV from LDS V^T tile
#pragma unroll
        for (int t = 0; t < 4; ++t) {
            v16bf bv = *(const v16bf*)&lds_v[buf][16 * t + r][16 * g];
            o[t] = wmma_bf16(ap, bv, o[t]);
        }
    }

    // ---- normalize + write attention output as bf16 tokens x 1024
    float inv[8];
#pragma unroll
    for (int rr = 0; rr < 8; ++rr) inv[rr] = 1.0f / l[rr];
    int b_ = bh >> 4, h = bh & 15;
#pragma unroll
    for (int t = 0; t < 4; ++t)
#pragma unroll
        for (int rr = 0; rr < 8; ++rr) {
            int srow = q0 + rr + 8 * g;
            int col = h * 64 + 16 * t + r;
            AO[(size_t)(b_ * 2048 + srow) * 1024 + col] = (bf16)(o[t][rr] * inv[rr]);
        }
}

// ---------------------------------------------------------------------------
// Kernel 3: output projection. out[8192,1024] = AO @ Wout + bias  (f32 out)
// Same async-LDS weight staging as kernel 1; A is already bf16.
// ---------------------------------------------------------------------------
__global__ __launch_bounds__(256) void k_out_gemm(
    const bf16* __restrict__ AO, const bf16* __restrict__ WT,
    const float* __restrict__ bias, float* __restrict__ out)
{
    __shared__ alignas(64) bf16 lds_b[2][64][32];

    const int tid = threadIdx.x;
    const int lane = tid & 31;
    const int wave = tid >> 5;
    const int g = lane >> 4, r = lane & 15;
    const int row0 = blockIdx.y * 128 + wave * 16;
    const int col0 = blockIdx.x * 64;

    const int bn = tid >> 2;
    const int bk = (tid & 3) * 8;
    const bf16* wsrc = WT + (size_t)(col0 + bn) * 1024 + bk;

    v8f acc[4];
#pragma unroll
    for (int t = 0; t < 4; ++t) acc[t] = zero8();

    const bf16* arow = AO + (size_t)(row0 + r) * 1024;

    async_b128(&lds_b[0][bn][bk], wsrc);
    for (int k0 = 0; k0 < 1024; k0 += 32) {
        const int buf = (k0 >> 5) & 1;
        wait_async0();
        __syncthreads();
        if (k0 + 32 < 1024) async_b128(&lds_b[buf ^ 1][bn][bk], wsrc + k0 + 32);

        __builtin_prefetch(arow + k0 + 64, 0, 1);
        v8bf lo = *(const v8bf*)(arow + k0 + 16 * g);
        v8bf hi = *(const v8bf*)(arow + k0 + 16 * g + 8);
        v16bf a;
#pragma unroll
        for (int i = 0; i < 8; ++i) { a[i] = lo[i]; a[8 + i] = hi[i]; }
#pragma unroll
        for (int t = 0; t < 4; ++t) {
            v16bf b = *(const v16bf*)&lds_b[buf][16 * t + r][16 * g];
            acc[t] = wmma_bf16(a, b, acc[t]);
        }
    }
#pragma unroll
    for (int t = 0; t < 4; ++t) {
        int col = col0 + 16 * t + r;
        float bv = bias[col];
#pragma unroll
        for (int rr = 0; rr < 8; ++rr) {
            int row = row0 + rr + 8 * g;
            out[(size_t)row * 1024 + col] = acc[t][rr] + bv;
        }
    }
}

// ---------------------------------------------------------------------------
extern "C" void kernel_launch(void* const* d_in, const int* in_sizes, int n_in,
                              void* d_out, int out_size, void* d_ws, size_t ws_size,
                              hipStream_t stream) {
    const float* x    = (const float*)d_in[0];   // [4,2048,1024]
    const float* Wqkv = (const float*)d_in[1];   // [1024,3072]
    const float* bqkv = (const float*)d_in[2];   // [3072]
    const float* Wout = (const float*)d_in[3];   // [1024,1024]
    const float* bout = (const float*)d_in[4];   // [1024]
    float* out = (float*)d_out;                  // [4,2048,1024]

    char* ws = (char*)d_ws;
    bf16* WTqkv = (bf16*)(ws);                                    //  6 MB [3072,1024]
    bf16* WTout = (bf16*)(ws + (size_t)6291456);                  //  2 MB [1024,1024]
    bf16* Qb    = (bf16*)(ws + (size_t)8388608);                  // 16 MB [64,2048,64]
    bf16* Kb    = (bf16*)(ws + (size_t)8388608 + 16777216);       // 16 MB [64,2048,64]
    bf16* VTb   = (bf16*)(ws + (size_t)8388608 + 2*16777216ull);  // 16 MB [64,64,2048]
    bf16* AO    = (bf16*)(ws + (size_t)8388608 + 3*16777216ull);  // 16 MB [8192,1024]

    k_transpose_bf16<<<(3072 * 1024 + 255) / 256, 256, 0, stream>>>(Wqkv, WTqkv, 1024, 3072);
    k_transpose_bf16<<<(1024 * 1024 + 255) / 256, 256, 0, stream>>>(Wout, WTout, 1024, 1024);
    k_qkv_gemm<<<dim3(48, 64), 256, 0, stream>>>(x, WTqkv, bqkv, Qb, Kb, VTb);
    k_attention<<<dim3(16, 64), 256, 0, stream>>>(Qb, Kb, VTb, AO);
    k_out_gemm<<<dim3(16, 64), 256, 0, stream>>>(AO, WTout, bout, out);
}
